// RNN2DGeneral_3332894621960
// MI455X (gfx1250) — matrix operand order repaired and checked by reading, not compiled, not run on hardware
//
#include <hip/hip_runtime.h>
#include <hip/hip_bf16.h>
#include <math.h>

typedef __attribute__((ext_vector_type(16))) _Float16 v16h;
typedef __attribute__((ext_vector_type(8)))  _Float16 v8h;
typedef __attribute__((ext_vector_type(8)))  float    v8f;

#define HD    128   // hidden size
#define LL    32    // lattice side
#define DEPTH 2
#define MTILE 16    // batch samples per workgroup (WMMA M)
#define NWAVES 8    // waves per WG; wave w owns hidden cols [16w,16w+16)

// LDS layout (dynamic):
//   vrow : [DEPTH][LL][MTILE][HD] f16  = 256 KB   (previous-row carries, per column)
//   hbuf : [2][DEPTH][MTILE][HD] f16   =  16 KB   (ping-pong horizontal carry / h0,h1)
//   lgts : [2][MTILE] f32              = 128 B    (double-buffered logit reduction)
//   stateB : 16 bytes (padded to 64)              (packed per-sample one-hot indices)
#define VROW_ELTS (DEPTH*LL*MTILE*HD)
#define HBUF_ELTS (2*DEPTH*MTILE*HD)
#define LDS_BYTES (VROW_ELTS*2 + HBUF_ELTS*2 + 2*MTILE*4 + 64)

__device__ __forceinline__ v8f wmma16(v16h a, v16h b, v8f c) {
  // D = A(16x32 f16) x B(32x16 f16) + C(16x16 f32)
  return __builtin_amdgcn_wmma_f32_16x16x32_f16(false, a, false, b, (short)0, c,
                                                false, false);
}

// A fragment (16x32 f16) from a row-major [MTILE][HD] f16 tile in LDS.
// Per ISA layout: lane m = l&15; K runs {0..7,16..23} (+8 for lanes 16..31).
__device__ __forceinline__ v16h ldsAFrag(const _Float16* tile, int lane, int kc) {
  const int m  = lane & 15;
  const int k0 = kc * 32 + ((lane >> 4) << 3);
  const v8h lo = *(const v8h*)(tile + m * HD + k0);
  const v8h hi = *(const v8h*)(tile + m * HD + k0 + 16);
  return __builtin_shufflevector(lo, hi, 0,1,2,3,4,5,6,7,8,9,10,11,12,13,14,15);
}

// B fragment (32x16 f16) gathered once from a global row-major [K=128][N=128] f32
// weight matrix. Per lane: column n, K runs {0..7,16..23} (+8 for lanes 16..31).
__device__ __forceinline__ v16h gblBFrag(const float* __restrict__ W,
                                         int lane, int kc, int n) {
  const int kb = kc * 32 + ((lane >> 4) << 3);
  v16h f;
#pragma unroll
  for (int j = 0; j < 16; ++j) f[j] = (_Float16)W[(kb + j + ((j >> 3) << 3)) * HD + n];
  return f;
}

__device__ __forceinline__ float eluf(float v) {
  return (v > 0.f) ? v : expm1f(v);
}

__global__ __launch_bounds__(NWAVES * 32, 1)
void rnn2d_wmma_kernel(const int*   __restrict__ x,
                       const float* __restrict__ WH,
                       const float* __restrict__ WV,
                       const float* __restrict__ WS0,
                       const float* __restrict__ WS1,
                       const float* __restrict__ Wout,
                       const float* __restrict__ bout,
                       float* __restrict__ out, int Bn)
{
  extern __shared__ char lds[];
  _Float16* vrow = (_Float16*)lds;                   // [DEPTH][LL][MTILE][HD]
  _Float16* hbuf = vrow + VROW_ELTS;                 // [2][DEPTH][MTILE][HD]
  float*    lgts = (float*)(hbuf + HBUF_ELTS);       // [2][MTILE]
  unsigned char* stateB = (unsigned char*)(lgts + 2*MTILE);  // [16]

  const int tid  = threadIdx.x;
  const int wave = tid >> 5;
  const int lane = tid & 31;
  const int nloc = lane & 15;
  const int nglb = wave * 16 + nloc;
  const int b0   = blockIdx.x * MTILE;

  // ---- one-time: resident weight B-fragments (160 VGPRs/wave) ----
  v16h wh0[4], wv0[4], wh1[4], wv1[4], ws1f[4];
#pragma unroll
  for (int kc = 0; kc < 4; ++kc) {
    wh0[kc]  = gblBFrag(WH,          lane, kc, nglb);
    wv0[kc]  = gblBFrag(WV,          lane, kc, nglb);
    wh1[kc]  = gblBFrag(WH + HD*HD,  lane, kc, nglb);
    wv1[kc]  = gblBFrag(WV + HD*HD,  lane, kc, nglb);
    ws1f[kc] = gblBFrag(WS1,         lane, kc, nglb);
  }
  float ws0n[4];
#pragma unroll
  for (int j = 0; j < 4; ++j) ws0n[j] = WS0[j * HD + nglb];
  const float woutn = Wout[nglb];
  const float b_out = bout[0];

  // ---- zero all LDS state ----
  {
    uint4 z = {0u,0u,0u,0u};
    uint4* p = (uint4*)lds;
    for (int i = tid; i < (int)(LDS_BYTES / 16); i += NWAVES * 32) p[i] = z;
  }
  __syncthreads();

  _Float16* hc = hbuf;                       // carries read this site
  _Float16* hn = hbuf + DEPTH * MTILE * HD;  // carries written this site

  float accOut = 0.0f;  // per-sample log-amplitude (wave 0, lanes 0..15)

  for (int r = 0; r < LL; ++r) {
    // horizontal carry resets to zero at each row start
    {
      uint4 z = {0u,0u,0u,0u};
      uint4* p = (uint4*)hc;
      for (int i = tid; i < (int)(DEPTH*MTILE*HD*2 / 16); i += NWAVES * 32) p[i] = z;
    }
    __syncthreads();

    const int rev = r & 1;  // boustrophedon direction
    for (int t = 0; t < LL; ++t) {
      const int c = rev ? (LL - 1 - t) : t;

      // wave 7 packs per-sample one-hot indices (xh,xv) for this site
      if (wave == 7 && lane < MTILE) {
        const int m = lane;
        int xh = 2, xv = 2;  // 2 == "no input" (zero one-hot)
        if (b0 + m < Bn) {
          const int base = (b0 + m) * (LL * LL);
          if (t) { const int cp = rev ? (c + 1) : (c - 1); xh = x[base + r*LL + cp] & 1; }
          if (r) { xv = x[base + (r - 1)*LL + c] & 1; }
        }
        stateB[m] = (unsigned char)(xh | (xv << 4));
      }

      _Float16* v0c = vrow + (0 * LL + c) * (MTILE * HD);
      _Float16* v1c = vrow + (1 * LL + c) * (MTILE * HD);

      // ---- layer 0: carryH0@WH0 + carryV0@WV0 ----
      v8f acc0 = {};
#pragma unroll
      for (int kc = 0; kc < 4; ++kc) {
        acc0 = wmma16(ldsAFrag(hc,  lane, kc), wh0[kc], acc0);
        acc0 = wmma16(ldsAFrag(v0c, lane, kc), wv0[kc], acc0);
      }
      __syncthreads();  // B1: stateB visible; all reads of hc[0]/v0c done

      // + state@WS0 (one-hot row select), ELU -> h0; store as next carries
      {
        uint4 sw = *(const uint4*)stateB;
        const unsigned int wlo = (lane < 16) ? sw.x : sw.z;
        const unsigned int whi = (lane < 16) ? sw.y : sw.w;
#pragma unroll
        for (int rr = 0; rr < 8; ++rr) {
          const unsigned int by = ((rr < 4 ? wlo : whi) >> ((rr & 3) * 8)) & 0xFFu;
          const int xh = by & 15, xv = (by >> 4) & 15;
          float v = acc0[rr];
          if (xh < 2) v += ws0n[xh];
          if (xv < 2) v += ws0n[2 + xv];
          v = eluf(v);
          const int m = rr + ((lane >> 4) << 3);
          const _Float16 hv = (_Float16)v;
          hn[m * HD + nglb]  = hv;   // h0: layer-1 input & next-site carryH0
          v0c[m * HD + nglb] = hv;   // next-row carryV0 at column c
        }
      }

      // ---- layer 1 partial: carryH1@WH1 + carryV1@WV1 ----
      v8f acc1 = {};
#pragma unroll
      for (int kc = 0; kc < 4; ++kc) {
        acc1 = wmma16(ldsAFrag(hc + MTILE * HD, lane, kc), wh1[kc], acc1);
        acc1 = wmma16(ldsAFrag(v1c,             lane, kc), wv1[kc], acc1);
      }
      __syncthreads();  // B2: h0 visible; all reads of hc[1]/v1c done

      // ---- + h0@WS1, ELU -> h1, store, logit partial reduction ----
#pragma unroll
      for (int kc = 0; kc < 4; ++kc)
        acc1 = wmma16(ldsAFrag(hn, lane, kc), ws1f[kc], acc1);

      float* lg = lgts + (t & 1) * MTILE;
#pragma unroll
      for (int rr = 0; rr < 8; ++rr) {
        float v = eluf(acc1[rr]);
        const int m = rr + ((lane >> 4) << 3);
        const _Float16 hv = (_Float16)v;
        hn[(MTILE + m) * HD + nglb] = hv;  // next-site carryH1
        v1c[m * HD + nglb]          = hv;  // next-row carryV1
        float p = v * woutn;               // h1 @ Wout partial
        p += __shfl_xor(p, 1, 32);
        p += __shfl_xor(p, 2, 32);
        p += __shfl_xor(p, 4, 32);
        p += __shfl_xor(p, 8, 32);
        if (nloc == 0) atomicAdd(&lg[m], p);
      }
      __syncthreads();  // B3: logits complete

      // ---- log-softmax site term (wave 0) ----
      if (wave == 0 && lane < MTILE && (b0 + lane) < Bn) {
        const int m = lane;
        const float tl = lg[m] + b_out;
        lg[m] = 0.f;  // reset buffer for reuse at site t+2
        const float lse = fmaxf(tl, 0.f) + log1pf(expf(-fabsf(tl)));
        const int xc = x[(b0 + m) * (LL * LL) + r * LL + c] & 1;
        float lp = 0.5f * ((xc ? tl : 0.f) - lse);
        if (!(lp == lp)) lp = -35.0f;  // nan_to_num
        accOut += lp;
      }

      { _Float16* tmp = hc; hc = hn; hn = tmp; }  // ping-pong carries
    }
  }

  if (wave == 0 && lane < MTILE && (b0 + lane) < Bn)
    out[b0 + lane] = accOut;
}

extern "C" void kernel_launch(void* const* d_in, const int* in_sizes, int n_in,
                              void* d_out, int out_size, void* d_ws, size_t ws_size,
                              hipStream_t stream) {
  const int*   x    = (const int*)  d_in[0];
  const float* WH   = (const float*)d_in[1];
  const float* WV   = (const float*)d_in[2];
  const float* WS0  = (const float*)d_in[3];
  const float* WS1  = (const float*)d_in[4];
  const float* Wout = (const float*)d_in[5];
  const float* bout = (const float*)d_in[6];
  float* out = (float*)d_out;

  const int B = in_sizes[0] / (LL * LL);           // 1024
  const int grid = (B + MTILE - 1) / MTILE;        // 64 workgroups

  rnn2d_wmma_kernel<<<grid, NWAVES * 32, LDS_BYTES, stream>>>(
      x, WH, WV, WS0, WS1, Wout, bout, out, B);
}
